// Encoder_20340965114371
// MI455X (gfx1250) — matrix operand order repaired and checked by reading, not compiled, not run on hardware
//
#include <hip/hip_runtime.h>
#include <hip/hip_bf16.h>

// ---------------------------------------------------------------------------
// CDNA5 (gfx1250) encoder pipeline: all matmuls via v_wmma_f32_16x16x32_f16.
//  - f16 token-major activation buffers, f32 accumulate everywhere
//  - K=3 convs as 3 shifted accumulating GEMMs over zero-padded token rows
//  - flash-style attention with per-wave 16-row tiles (wave32 layouts)
//  - GEMM: 32x32 per wave (2x2 frags), K-tile 64, double-buffered LDS,
//    one barrier per tile, global loads overlapped with WMMAs
// ---------------------------------------------------------------------------

typedef __attribute__((ext_vector_type(16))) _Float16 v16h;
typedef __attribute__((ext_vector_type(8)))  float    v8f;

#define BB   8
#define CC   512
#define TT   1024
#define FF   2048
#define NH   8
#define DH   64
#define TP   (TT + 2)          // padded tokens per batch (zero row at t=-1 and t=T)
#define MREAL (BB * TP)        // 8208
#define MPA  8256              // MREAL rounded up to 64 (GEMM M-tile)
#define BHH  (BB * NH)         // 64
#define NL   6

__device__ __forceinline__ v8f wmma_f16(v16h a, v16h b, v8f c) {
  return __builtin_amdgcn_wmma_f32_16x16x32_f16(false, a, false, b, (short)0, c,
                                                false, false);
}

// 16-bit A-matrix 16x32 k index for frag slot e, lane-half: contiguous k = 16*half + e.
__device__ __forceinline__ int afrag_k(int e, int half) {
  return e + (half ? 16 : 0);
}

// ---------------------------------------------------------------------------
// WMMA GEMM:  C[M,N] (f32) (+)= A[M,K] (f16) * B[N,K]^T (f16) + bias
// grid = (N/128, M/64), block = 256 (8 waves as 2(M) x 4(N), 32x32 per wave).
// K-tile 64, ping-pong LDS: next tile's global loads issue before this tile's
// 8 WMMAs; stores land in the alternate buffer; one barrier per tile.
// ---------------------------------------------------------------------------
__global__ __launch_bounds__(256)
void k_gemm_nt(const _Float16* __restrict__ A, const _Float16* __restrict__ Bm,
               float* __restrict__ Cm, const float* __restrict__ bias,
               int N, int K, int accumulate)
{
  __shared__ __align__(32) _Float16 As[2][64 * 64];    // 2 x 8 KB
  __shared__ __align__(32) _Float16 Bs[2][128 * 64];   // 2 x 16 KB

  const int tid  = threadIdx.x;
  const int wave = tid >> 5, lane = tid & 31;
  const int wm   = wave & 1, wn = wave >> 1;        // 2(M) x 4(N) waves
  const int half = lane >> 4, l15 = lane & 15;
  const int bm   = blockIdx.y * 64, bn = blockIdx.x * 128;
  const int m0   = wm * 32, n0 = wn * 32;

  const int arow = tid >> 2;          // 0..63
  const int acol = (tid & 3) * 16;    // 0,16,32,48
  const int brow = tid >> 1;          // 0..127
  const int bcol = (tid & 1) * 32;    // 0,32

  uint4 ra0, ra1, rb0, rb1, rb2, rb3;          // staging registers
  auto gload = [&](int k0) {
    const _Float16* ga = A + (size_t)(bm + arow) * K + (k0 + acol);
    ra0 = *(const uint4*)ga;
    ra1 = *(const uint4*)(ga + 8);
    const _Float16* gb = Bm + (size_t)(bn + brow) * K + (k0 + bcol);
    rb0 = *(const uint4*)gb;
    rb1 = *(const uint4*)(gb + 8);
    rb2 = *(const uint4*)(gb + 16);
    rb3 = *(const uint4*)(gb + 24);
    if (k0 + 64 < K) {                         // emits global_prefetch_b8
      __builtin_prefetch(ga + 64, 0, 3);
      __builtin_prefetch(gb + 64, 0, 3);
    }
  };
  auto lstore = [&](int buf) {
    *(uint4*)&As[buf][arow * 64 + acol]      = ra0;
    *(uint4*)&As[buf][arow * 64 + acol + 8]  = ra1;
    *(uint4*)&Bs[buf][brow * 64 + bcol]      = rb0;
    *(uint4*)&Bs[buf][brow * 64 + bcol + 8]  = rb1;
    *(uint4*)&Bs[buf][brow * 64 + bcol + 16] = rb2;
    *(uint4*)&Bs[buf][brow * 64 + bcol + 24] = rb3;
  };

  v8f acc[2][2] = {};

  gload(0);
  lstore(0);
  __syncthreads();

  int cur = 0;
  for (int k0 = 0; k0 < K; k0 += 64, cur ^= 1) {
    const bool more = (k0 + 64 < K);
    if (more) gload(k0 + 64);                  // in flight during the WMMAs

#pragma unroll
    for (int ks = 0; ks < 2; ++ks) {           // two 32-K sub-steps
      v16h a0, a1;
      {
        const _Float16* ap0 = &As[cur][(m0 + l15) * 64 + ks * 32];
        const _Float16* ap1 = &As[cur][(m0 + 16 + l15) * 64 + ks * 32];
#pragma unroll
        for (int e = 0; e < 16; ++e) {
          int k = afrag_k(e, half);
          a0[e] = ap0[k];
          a1[e] = ap1[k];
        }
      }
      v16h b0 = *(const v16h*)&Bs[cur][(n0 + l15) * 64 + ks * 32 + half * 16];
      v16h b1 = *(const v16h*)&Bs[cur][(n0 + 16 + l15) * 64 + ks * 32 + half * 16];
      acc[0][0] = wmma_f16(a0, b0, acc[0][0]);
      acc[0][1] = wmma_f16(a0, b1, acc[0][1]);
      acc[1][0] = wmma_f16(a1, b0, acc[1][0]);
      acc[1][1] = wmma_f16(a1, b1, acc[1][1]);
    }

    if (more) lstore(cur ^ 1);                 // alternate buffer, no race
    __syncthreads();                           // one barrier per K-tile
  }

  const int col0 = bn + n0 + l15;
  const float bi0 = bias ? bias[col0]      : 0.f;
  const float bi1 = bias ? bias[col0 + 16] : 0.f;
#pragma unroll
  for (int mi = 0; mi < 2; ++mi) {
    const int row0 = bm + m0 + mi * 16 + half * 8;
#pragma unroll
    for (int r = 0; r < 8; ++r) {
      size_t i0 = (size_t)(row0 + r) * N + col0;
      float v0 = acc[mi][0][r] + bi0;
      float v1 = acc[mi][1][r] + bi1;
      if (accumulate) { v0 += Cm[i0]; v1 += Cm[i0 + 16]; }
      Cm[i0] = v0; Cm[i0 + 16] = v1;
    }
  }
}

// ---------------------------------------------------------------------------
// Flash attention: grid = BH*(T/64), block = 128 (4 waves; 16 t-rows each).
// S = Qs*K^T (+rel-K bias in |s-t|<=4 band, mask), online softmax, O = P*V.
// Row stats live per-lane: C-frag row r+8*half <-> stat register r.
// ---------------------------------------------------------------------------
__global__ __launch_bounds__(128)
void k_flash(const _Float16* __restrict__ Qh, const _Float16* __restrict__ Kh,
             const _Float16* __restrict__ Vh, const float* __restrict__ Rb,
             const float* __restrict__ xm, float* __restrict__ O,
             float* __restrict__ rowM, float* __restrict__ rowL)
{
  __shared__ __align__(32) _Float16 Ks[32 * DH];
  __shared__ __align__(32) _Float16 Vs[32 * DH];
  __shared__ __align__(32) _Float16 Pt[4][16 * 32];

  const int bh   = blockIdx.x >> 4;      // T/64 == 16
  const int tblk = blockIdx.x & 15;
  const int b    = bh >> 3;              // NH == 8
  const int tid  = threadIdx.x;
  const int wave = tid >> 5, lane = tid & 31;
  const int half = lane >> 4, l15 = lane & 15;
  const int t0   = tblk * 64 + wave * 16;

  v16h q0, q1;                           // Q frags (D=64 -> two 16x32)
  {
    const _Float16* qr = Qh + ((size_t)bh * TT + t0 + l15) * DH;
#pragma unroll
    for (int e = 0; e < 16; ++e) {
      int k = afrag_k(e, half);
      q0[e] = qr[k];
      q1[e] = qr[32 + k];
    }
  }
  float mtk[8];
#pragma unroll
  for (int r = 0; r < 8; ++r) mtk[r] = xm[b * TT + t0 + r + half * 8];

  float mrun[8], lrun[8];
#pragma unroll
  for (int r = 0; r < 8; ++r) { mrun[r] = -1e30f; lrun[r] = 0.f; }
  v8f oacc[4] = {};

  for (int s0 = 0; s0 < TT; s0 += 32) {
    {                                    // stage 32x64 K and V chunks
      const int rr = tid >> 2;
      const int c0 = (tid & 3) * 16;
      const _Float16* kg = Kh + ((size_t)bh * TT + s0 + rr) * DH + c0;
      const _Float16* vg = Vh + ((size_t)bh * TT + s0 + rr) * DH + c0;
      uint4 k0r = *(const uint4*)kg;
      uint4 k1r = *(const uint4*)(kg + 8);
      uint4 v0r = *(const uint4*)vg;
      uint4 v1r = *(const uint4*)(vg + 8);
      *(uint4*)&Ks[rr * DH + c0]     = k0r;
      *(uint4*)&Ks[rr * DH + c0 + 8] = k1r;
      *(uint4*)&Vs[rr * DH + c0]     = v0r;
      *(uint4*)&Vs[rr * DH + c0 + 8] = v1r;
    }
    __syncthreads();

    v8f sA = {}, sB = {};
    {
      const _Float16* k0p = &Ks[l15 * DH + half * 16];
      const _Float16* k1p = &Ks[(16 + l15) * DH + half * 16];
      sA = wmma_f16(q0, *(const v16h*)k0p,        sA);
      sA = wmma_f16(q1, *(const v16h*)(k0p + 32), sA);
      sB = wmma_f16(q0, *(const v16h*)k1p,        sB);
      sB = wmma_f16(q1, *(const v16h*)(k1p + 32), sB);
    }

    const int sa = s0 + l15, sb = sa + 16;
    const float mska = xm[b * TT + sa];
    const float mskb = xm[b * TT + sb];
#pragma unroll
    for (int r = 0; r < 8; ++r) {
      const int trow = t0 + r + half * 8;
      const float* Rr = Rb + ((size_t)bh * TT + trow) * 9;
      float va = sA[r], vb = sB[r];
      const int da = sa - trow, db = sb - trow;
      if (da >= -4 && da <= 4) va += Rr[da + 4];
      if (db >= -4 && db <= 4) vb += Rr[db + 4];
      if (mska * mtk[r] == 0.f) va = -1e4f;
      if (mskb * mtk[r] == 0.f) vb = -1e4f;
      float mx = fmaxf(va, vb);
#pragma unroll
      for (int i = 1; i < 16; i <<= 1) mx = fmaxf(mx, __shfl_xor(mx, i, 32));
      const float mnew = fmaxf(mrun[r], mx);
      const float fac  = __expf(mrun[r] - mnew);
      const float pa   = __expf(va - mnew);
      const float pb   = __expf(vb - mnew);
      float ps = pa + pb;
#pragma unroll
      for (int i = 1; i < 16; i <<= 1) ps += __shfl_xor(ps, i, 32);
      lrun[r] = lrun[r] * fac + ps;
      mrun[r] = mnew;
#pragma unroll
      for (int d = 0; d < 4; ++d) oacc[d][r] *= fac;
      Pt[wave][(r + half * 8) * 32 + l15]      = (_Float16)pa;
      Pt[wave][(r + half * 8) * 32 + 16 + l15] = (_Float16)pb;
    }
    __syncthreads();

    v16h pf;
    {
      const _Float16* pp = &Pt[wave][l15 * 32];
#pragma unroll
      for (int e = 0; e < 16; ++e) pf[e] = pp[afrag_k(e, half)];
    }
#pragma unroll
    for (int dsub = 0; dsub < 4; ++dsub) {
      v16h vf;
#pragma unroll
      for (int e = 0; e < 16; ++e)
        vf[e] = Vs[(half * 16 + e) * DH + dsub * 16 + l15];
      oacc[dsub] = wmma_f16(pf, vf, oacc[dsub]);
    }
    __syncthreads();
  }

#pragma unroll
  for (int r = 0; r < 8; ++r) {
    const float inv = 1.f / lrun[r];
    const size_t base = ((size_t)bh * TT + t0 + r + half * 8) * DH + l15;
    O[base]      = oacc[0][r] * inv;
    O[base + 16] = oacc[1][r] * inv;
    O[base + 32] = oacc[2][r] * inv;
    O[base + 48] = oacc[3][r] * inv;
  }
  if (l15 == 0) {
#pragma unroll
    for (int r = 0; r < 8; ++r) {
      rowM[(size_t)bh * TT + t0 + r + half * 8] = mrun[r];
      rowL[(size_t)bh * TT + t0 + r + half * 8] = lrun[r];
    }
  }
}

// --------------------------- small support kernels -------------------------

__global__ void k_zero16(_Float16* p, size_t n) {
  size_t i = (size_t)blockIdx.x * blockDim.x + threadIdx.x;
  const size_t stride = (size_t)gridDim.x * blockDim.x;
  for (; i < n; i += stride) p[i] = (_Float16)0.f;
}

__global__ void k_cvt(const float* __restrict__ s, _Float16* __restrict__ d, size_t n) {
  size_t i = (size_t)blockIdx.x * blockDim.x + threadIdx.x;
  const size_t stride = (size_t)gridDim.x * blockDim.x;
  for (; i < n; i += stride) d[i] = (_Float16)s[i];
}

// ffn_w1 [L][F][C][3] -> f16 [L][3][F][C]
__global__ void k_cvt_w1(const float* __restrict__ s, _Float16* __restrict__ d) {
  const size_t n = (size_t)NL * FF * CC * 3;
  size_t i = (size_t)blockIdx.x * blockDim.x + threadIdx.x;
  const size_t stride = (size_t)gridDim.x * blockDim.x;
  for (; i < n; i += stride) {
    size_t l = i / ((size_t)FF * CC * 3);
    size_t r = i % ((size_t)FF * CC * 3);
    int f  = (int)(r / (CC * 3));
    int r2 = (int)(r % (CC * 3));
    int c = r2 / 3, dt = r2 % 3;
    d[l * 3 * FF * CC + ((size_t)dt * FF + f) * CC + c] = (_Float16)s[i];
  }
}

// ffn_w2 [L][C][F][3] -> f16 [L][3][C][F]
__global__ void k_cvt_w2(const float* __restrict__ s, _Float16* __restrict__ d) {
  const size_t n = (size_t)NL * CC * FF * 3;
  size_t i = (size_t)blockIdx.x * blockDim.x + threadIdx.x;
  const size_t stride = (size_t)gridDim.x * blockDim.x;
  for (; i < n; i += stride) {
    size_t l = i / ((size_t)CC * FF * 3);
    size_t r = i % ((size_t)CC * FF * 3);
    int o  = (int)(r / (FF * 3));
    int r2 = (int)(r % (FF * 3));
    int f = r2 / 3, dt = r2 % 3;
    d[l * 3 * CC * FF + ((size_t)dt * CC + o) * FF + f] = (_Float16)s[i];
  }
}

// x*mask -> X f32 [B,C,T] and padded token-major f16 [row, c]
__global__ void k_x0pack(const float* __restrict__ x, const float* __restrict__ xm,
                         float* __restrict__ X, _Float16* __restrict__ Xtok) {
  int idx = blockIdx.x * blockDim.x + threadIdx.x;   // B*C*T
  int t = idx % TT;
  int bc = idx / TT;
  int c = bc % CC, b = bc / CC;
  float v = x[idx] * xm[b * TT + t];
  X[idx] = v;
  Xtok[((size_t)b * TP + 1 + t) * CC + c] = (_Float16)v;
}

// QKV f32 [row,1536] -> Qh(scaled)/Kh/Vh f16 [bh,t,d]
__global__ void k_qkvpack(const float* __restrict__ QKV, _Float16* __restrict__ Qh,
                          _Float16* __restrict__ Kh, _Float16* __restrict__ Vh) {
  int idx = blockIdx.x * blockDim.x + threadIdx.x;   // B*T*C
  int c = idx & (CC - 1);
  int bt = idx >> 9;
  int t = bt % TT, b = bt / TT;
  int h = c >> 6, d = c & 63;
  const float* src = QKV + ((size_t)b * TP + 1 + t) * (3 * CC);
  size_t o = (((size_t)(b * NH + h)) * TT + t) * DH + d;
  Qh[o] = (_Float16)(src[c] * 0.125f);               // 1/sqrt(64)
  Kh[o] = (_Float16)src[CC + c];
  Vh[o] = (_Float16)src[2 * CC + c];
}

// R[bh,t,off] = Qs[t] . erk[off]   (off = s-t+4 in [0,8])
__global__ void k_relbias(const _Float16* __restrict__ Qh, const float* __restrict__ erk,
                          float* __restrict__ Rb) {
  int idx = blockIdx.x * blockDim.x + threadIdx.x;   // BH*T
  int t = idx % TT;
  const _Float16* qp = Qh + (size_t)idx * DH;
  float q[DH];
#pragma unroll
  for (int d = 0; d < DH; ++d) q[d] = (float)qp[d];
#pragma unroll
  for (int off = 0; off < 9; ++off) {
    int s2 = t + off - 4;
    float v = 0.f;
    if (s2 >= 0 && s2 < TT) {
      const float* er = erk + off * DH;
#pragma unroll
      for (int d = 0; d < DH; ++d) v += q[d] * er[d];
    }
    Rb[(size_t)idx * 9 + off] = v;
  }
}

// add relative-V term: O[t] += sum_off p[t,t+off-4] * erv[off]
__global__ void k_bandv(const _Float16* __restrict__ Qh, const _Float16* __restrict__ Kh,
                        const float* __restrict__ Rb, const float* __restrict__ erv,
                        const float* __restrict__ xm, const float* __restrict__ rowM,
                        const float* __restrict__ rowL, float* __restrict__ O) {
  int idx = blockIdx.x * blockDim.x + threadIdx.x;   // BH*T
  int bh = idx / TT, t = idx % TT;
  int b = bh >> 3;
  const _Float16* qp = Qh + (size_t)idx * DH;
  float q[DH];
#pragma unroll
  for (int d = 0; d < DH; ++d) q[d] = (float)qp[d];
  const float m  = rowM[idx];
  const float li = 1.f / rowL[idx];
  const float mt = xm[b * TT + t];
  float acc[DH];
#pragma unroll
  for (int d = 0; d < DH; ++d) acc[d] = 0.f;
  for (int off = 0; off < 9; ++off) {
    int s2 = t + off - 4;
    if (s2 < 0 || s2 >= TT) continue;
    const _Float16* kp = Kh + ((size_t)bh * TT + s2) * DH;
    float dot = 0.f;
#pragma unroll
    for (int d = 0; d < DH; ++d) dot += q[d] * (float)kp[d];
    dot += Rb[(size_t)idx * 9 + off];
    if (xm[b * TT + s2] * mt == 0.f) dot = -1e4f;
    const float p = __expf(dot - m) * li;
    const float* ev = erv + off * DH;
#pragma unroll
    for (int d = 0; d < DH; ++d) acc[d] += p * ev[d];
  }
  float* op = O + (size_t)idx * DH;
#pragma unroll
  for (int d = 0; d < DH; ++d) op[d] += acc[d];
}

// O f32 [bh,t,d] -> padded token-major f16
__global__ void k_attnpack(const float* __restrict__ O, _Float16* __restrict__ Atok) {
  int idx = blockIdx.x * blockDim.x + threadIdx.x;   // B*T*C
  int c = idx & (CC - 1);
  int bt = idx >> 9;
  int t = bt % TT, b = bt / TT;
  int h = c >> 6, d = c & 63;
  Atok[((size_t)b * TP + 1 + t) * CC + c] =
      (_Float16)O[(((size_t)(b * NH + h)) * TT + t) * DH + d];
}

// residual + channel LayerNorm; one wave per token (16 channels / lane)
__global__ __launch_bounds__(256)
void k_ln(const float* __restrict__ Y, const float* __restrict__ g,
          const float* __restrict__ bta, const float* __restrict__ xm,
          float* __restrict__ X, _Float16* __restrict__ Xtok, int mulmask)
{
  const int token = blockIdx.x * 8 + (threadIdx.x >> 5);
  const int lane  = threadIdx.x & 31;
  const int b = token / TT, t = token % TT;
  const size_t row = (size_t)b * TP + 1 + t;
  const float mk = xm[b * TT + t];
  float vals[16];
  float s = 0.f, s2 = 0.f;
#pragma unroll
  for (int i = 0; i < 16; ++i) {
    int c = lane + i * 32;
    float yv = Y[row * CC + c];
    if (mulmask) yv *= mk;
    float v = X[((size_t)b * CC + c) * TT + t] + yv;
    vals[i] = v; s += v; s2 += v * v;
  }
#pragma unroll
  for (int i = 1; i < 32; i <<= 1) {
    s  += __shfl_xor(s, i, 32);
    s2 += __shfl_xor(s2, i, 32);
  }
  const float mean = s * (1.f / CC);
  const float var  = s2 * (1.f / CC) - mean * mean;
  const float rs   = rsqrtf(var + 1e-5f);
#pragma unroll
  for (int i = 0; i < 16; ++i) {
    int c = lane + i * 32;
    float v = (vals[i] - mean) * rs * g[c] + bta[c];
    X[((size_t)b * CC + c) * TT + t] = v;
    Xtok[row * CC + c] = (_Float16)(v * mk);
  }
}

// relu + mask -> f16 hidden tokens
__global__ void k_relupack(const float* __restrict__ Hf, const float* __restrict__ xm,
                           _Float16* __restrict__ Htok) {
  int idx = blockIdx.x * blockDim.x + threadIdx.x;   // B*T*F
  int f = idx % FF;
  int bt = idx / FF;
  int t = bt % TT, b = bt / TT;
  size_t row = (size_t)b * TP + 1 + t;
  float v = fmaxf(Hf[row * FF + f], 0.f) * xm[b * TT + t];
  Htok[row * FF + f] = (_Float16)v;
}

__global__ void k_final(const float* __restrict__ X, const float* __restrict__ xm,
                        float* __restrict__ out) {
  int idx = blockIdx.x * blockDim.x + threadIdx.x;   // B*C*T
  int t = idx % TT;
  int b = idx / (CC * TT);
  out[idx] = X[idx] * xm[b * TT + t];
}

// ---------------------------------------------------------------------------

extern "C" void kernel_launch(void* const* d_in, const int* in_sizes, int n_in,
                              void* d_out, int out_size, void* d_ws, size_t ws_size,
                              hipStream_t stream) {
  (void)in_sizes; (void)n_in; (void)out_size; (void)ws_size;

  const float* x      = (const float*)d_in[0];
  const float* xm     = (const float*)d_in[1];
  const float* attn_w = (const float*)d_in[2];
  const float* attn_b = (const float*)d_in[3];
  const float* erk    = (const float*)d_in[4];
  const float* erv    = (const float*)d_in[5];
  const float* ln1g   = (const float*)d_in[6];
  const float* ln1b   = (const float*)d_in[7];
  const float* w1     = (const float*)d_in[8];
  const float* b1     = (const float*)d_in[9];
  const float* w2     = (const float*)d_in[10];
  const float* b2     = (const float*)d_in[11];
  const float* ln2g   = (const float*)d_in[12];
  const float* ln2b   = (const float*)d_in[13];

  char* ws = (char*)d_ws;
  size_t off = 0;
  auto alloc = [&](size_t bytes) -> char* {
    char* p = ws + off;
    off += (bytes + 255) & ~(size_t)255;
    return p;
  };

  float*     X     = (float*)alloc((size_t)BB * CC * TT * 4);
  _Float16*  XtokB = (_Float16*)alloc((size_t)(MPA + 2) * CC * 2);
  _Float16*  AtokB = (_Float16*)alloc((size_t)(MPA + 2) * CC * 2);
  _Float16*  HtokB = (_Float16*)alloc((size_t)(MPA + 2) * FF * 2);
  float*     QKV   = (float*)alloc((size_t)MPA * 3 * CC * 4);
  float*     Hf    = (float*)alloc((size_t)MPA * FF * 4);
  float*     Y     = (float*)alloc((size_t)MPA * CC * 4);
  _Float16*  Qh    = (_Float16*)alloc((size_t)BHH * TT * DH * 2);
  _Float16*  Kh    = (_Float16*)alloc((size_t)BHH * TT * DH * 2);
  _Float16*  Vh    = (_Float16*)alloc((size_t)BHH * TT * DH * 2);
  float*     O     = (float*)alloc((size_t)BHH * TT * DH * 4);
  float*     Rb    = (float*)alloc((size_t)BHH * TT * 9 * 4);
  float*     rm    = (float*)alloc((size_t)BHH * TT * 4);
  float*     rl    = (float*)alloc((size_t)BHH * TT * 4);
  _Float16*  Wattn = (_Float16*)alloc((size_t)NL * 4 * CC * CC * 2);
  _Float16*  W1c   = (_Float16*)alloc((size_t)NL * 3 * FF * CC * 2);
  _Float16*  W2c   = (_Float16*)alloc((size_t)NL * 3 * CC * FF * 2);

  _Float16* Xtok = XtokB + CC;   // base at row 0; guard row at -1 and MPA
  _Float16* Atok = AtokB + CC;
  _Float16* Htok = HtokB + FF;

  // per-call weight conversion + pad zeroing (idempotent, deterministic)
  k_cvt   <<<4096, 256, 0, stream>>>(attn_w, Wattn, (size_t)NL * 4 * CC * CC);
  k_cvt_w1<<<8192, 256, 0, stream>>>(w1, W1c);
  k_cvt_w2<<<8192, 256, 0, stream>>>(w2, W2c);
  k_zero16<<<2048, 256, 0, stream>>>(XtokB, (size_t)(MPA + 2) * CC);
  k_zero16<<<2048, 256, 0, stream>>>(AtokB, (size_t)(MPA + 2) * CC);
  k_zero16<<<4096, 256, 0, stream>>>(HtokB, (size_t)(MPA + 2) * FF);
  k_x0pack<<<(BB * CC * TT) / 256, 256, 0, stream>>>(x, xm, X, Xtok);

  for (int l = 0; l < NL; ++l) {
    const _Float16* Wa = Wattn + (size_t)l * 4 * CC * CC;
    const float*    ab = attn_b + (size_t)l * 4 * CC;

    // ---- attention ----
    k_gemm_nt<<<dim3(3 * CC / 128, MPA / 64), 256, 0, stream>>>(
        Xtok, Wa, QKV, ab, 3 * CC, CC, 0);
    k_qkvpack<<<(BB * TT * CC) / 256, 256, 0, stream>>>(QKV, Qh, Kh, Vh);
    k_relbias<<<(BHH * TT) / 256, 256, 0, stream>>>(Qh, erk + (size_t)l * 9 * DH, Rb);
    k_flash<<<BHH * (TT / 64), 128, 0, stream>>>(Qh, Kh, Vh, Rb, xm, O, rm, rl);
    k_bandv<<<(BHH * TT) / 256, 256, 0, stream>>>(
        Qh, Kh, Rb, erv + (size_t)l * 9 * DH, xm, rm, rl, O);
    k_attnpack<<<(BB * TT * CC) / 256, 256, 0, stream>>>(O, Atok);
    k_gemm_nt<<<dim3(CC / 128, MPA / 64), 256, 0, stream>>>(
        Atok, Wa + (size_t)3 * CC * CC, Y, ab + 3 * CC, CC, CC, 0);
    k_ln<<<BB * TT / 8, 256, 0, stream>>>(Y, ln1g + l * CC, ln1b + l * CC, xm, X, Xtok, 0);

    // ---- FFN: conv1 (K=3) as 3 shifted GEMMs ----
    for (int dt = 0; dt < 3; ++dt)
      k_gemm_nt<<<dim3(FF / 128, MPA / 64), 256, 0, stream>>>(
          Xtok + (dt - 1) * CC, W1c + ((size_t)l * 3 + dt) * FF * CC, Hf,
          dt == 0 ? b1 + (size_t)l * FF : nullptr, FF, CC, dt > 0);
    k_relupack<<<(BB * TT * FF) / 256, 256, 0, stream>>>(Hf, xm, Htok);
    for (int dt = 0; dt < 3; ++dt)
      k_gemm_nt<<<dim3(CC / 128, MPA / 64), 256, 0, stream>>>(
          Htok + (dt - 1) * FF, W2c + ((size_t)l * 3 + dt) * CC * FF, Y,
          dt == 0 ? b2 + (size_t)l * CC : nullptr, CC, FF, dt > 0);
    k_ln<<<BB * TT / 8, 256, 0, stream>>>(Y, ln2g + l * CC, ln2b + l * CC, xm, X, Xtok, 1);
  }

  k_final<<<(BB * CC * TT) / 256, 256, 0, stream>>>(X, xm, (float*)d_out);
}
